// PyramidAttention_75814762709541
// MI455X (gfx1250) — matrix operand order repaired and checked by reading, not compiled, not run on hardware
//
#include <hip/hip_runtime.h>

typedef __attribute__((ext_vector_type(16))) __bf16 v16bf;
typedef __attribute__((ext_vector_type(8)))  float  v8f;

namespace {
constexpr int kH = 48, kWd = 48, kHW = kH * kWd;
constexpr int kC = 64, kC2 = 32, kNB = 4;
constexpr int kL  = 7470;          // 48^2+43^2+38^2+33^2+28^2
constexpr int kLp = 7488;          // padded to multiple of 32
constexpr int kK  = 288;           // 32 * 3 * 3
constexpr int kHaloW = 50, kHalo = kHaloW * kHaloW;
constexpr float kSm = 10.0f;       // SOFTMAX_SCALE
}

__device__ __forceinline__ unsigned short f2bf(float f) {
  unsigned int u = __float_as_uint(f);
  unsigned int r = u + 0x7FFFu + ((u >> 16) & 1u);   // round-to-nearest-even
  return (unsigned short)(r >> 16);
}

__device__ __forceinline__ float cubicw(float x) {
  const float A = -0.75f;
  x = fabsf(x);
  if (x <= 1.f) return ((A + 2.f) * x - (A + 3.f)) * x * x + 1.f;
  if (x < 2.f)  return ((A * x - 5.f * A) * x + 8.f * A) * x - 4.f * A;
  return 0.f;
}

// ---- bicubic resize matrices: M_s[n_out, 48], one matrix per scale (H==W) ----
__global__ void build_resize_mats(float* __restrict__ rm) {
  const float scales[4] = {0.9f, 0.8f, 0.7f, 0.6f};
  const int   nout[4]   = {43, 38, 33, 28};
  int s = blockIdx.y;
  int i = blockIdx.x * blockDim.x + threadIdx.x;
  if (i >= nout[s]) return;
  float* M = rm + s * (43 * 48) + i * 48;
  for (int k = 0; k < 48; ++k) M[k] = 0.f;
  float src  = (i + 0.5f) / scales[s] - 0.5f;
  float ff   = floorf(src);
  float frac = src - ff;
  int   fi   = (int)ff;
  for (int k = -1; k < 3; ++k) {
    int idx = fi + k;
    idx = idx < 0 ? 0 : (idx > 47 ? 47 : idx);
    M[idx] += cubicw((float)k - frac);
  }
}

// tmp[c,o,w] = sum_h M[o,h] * x[c,h,w]
__global__ void resize_rows(const float* __restrict__ x, const float* __restrict__ M,
                            float* __restrict__ tmp, int n) {
  int t = blockIdx.x * blockDim.x + threadIdx.x;
  int total = kC * n * 48;
  if (t >= total) return;
  int w = t % 48, o = (t / 48) % n, c = t / (48 * n);
  const float* Mr = M + o * 48;
  const float* xc = x + c * kHW;
  float acc = 0.f;
  for (int h = 0; h < 48; ++h) acc += Mr[h] * xc[h * 48 + w];
  tmp[(c * n + o) * 48 + w] = acc;
}

// ref[c,o,p] = sum_w M[p,w] * tmp[c,o,w]
__global__ void resize_cols(const float* __restrict__ tmp, const float* __restrict__ M,
                            float* __restrict__ ref, int n) {
  int t = blockIdx.x * blockDim.x + threadIdx.x;
  int total = kC * n * n;
  if (t >= total) return;
  int p = t % n, o = (t / n) % n, c = t / (n * n);
  const float* Mr = M + p * 48;
  const float* tc = tmp + (c * n + o) * 48;
  float acc = 0.f;
  for (int w = 0; w < 48; ++w) acc += Mr[w] * tc[w];
  ref[(c * n + o) * n + p] = acc;
}

// out[co,p] = prelu(sum_ci w[co,ci]*in[ci,p] + b[co])
__global__ void conv1x1_prelu(const float* __restrict__ in, const float* __restrict__ w,
                              const float* __restrict__ b, const float* __restrict__ alpha,
                              float* __restrict__ out, int cin, int cout, int npix) {
  int t = blockIdx.x * blockDim.x + threadIdx.x;
  if (t >= cout * npix) return;
  int p = t % npix, co = t / npix;
  const float* wr = w + co * cin;
  float acc = b[co];
  for (int ci = 0; ci < cin; ++ci) acc += wr[ci] * in[ci * npix + p];
  float a = alpha[0];
  out[co * npix + p] = acc >= 0.f ? acc : a * acc;
}

// im2col of match_base -> bf16 [288, HW]
__global__ void build_pmb(const float* __restrict__ mb, unsigned short* __restrict__ pmb) {
  int t = blockIdx.x * blockDim.x + threadIdx.x;
  if (t >= kK * kHW) return;
  int p = t % kHW, k = t / kHW;
  int c = k / 9, ij = k % 9, i = ij / 3, j = ij % 3;
  int y = p / kWd, x = p % kWd;
  int yy = y + i - 1, xx = x + j - 1;
  float v = (yy >= 0 && yy < kH && xx >= 0 && xx < kWd) ? mb[(c * kH + yy) * kWd + xx] : 0.f;
  pmb[k * kHW + p] = f2bf(v);
}

// QK filters wi[l, 288] (f32, pre-norm) from feat_m patches for one scale
__global__ void build_wi(const float* __restrict__ fm, float* __restrict__ wiF,
                         int n, int loff) {
  int t = blockIdx.x * blockDim.x + threadIdx.x;
  if (t >= n * n * kK) return;
  int k = t % kK, l = t / kK;
  int c = k / 9, ij = k % 9, i = ij / 3, j = ij % 3;
  int o = l / n, p = l % n;
  int yy = o + i - 1, xx = p + j - 1;
  float v = (yy >= 0 && yy < n && xx >= 0 && xx < n) ? fm[(c * n + yy) * n + xx] : 0.f;
  wiF[(loff + l) * kK + k] = v;
}

// Rt[ij][c][l] = feat_a[c, o+1-i, p+1-j] (flipped/transposed value patches), bf16
__global__ void build_rt(const float* __restrict__ fa, unsigned short* __restrict__ Rt,
                         int n, int loff) {
  int t = blockIdx.x * blockDim.x + threadIdx.x;
  int nn = n * n;
  if (t >= 9 * kC * nn) return;
  int l = t % nn, c = (t / nn) % kC, ij = t / (nn * kC);
  int i = ij / 3, j = ij % 3;
  int o = l / n, p = l % n;
  int yy = o + 1 - i, xx = p + 1 - j;
  float v = (yy >= 0 && yy < n && xx >= 0 && xx < n) ? fa[(c * n + yy) * n + xx] : 0.f;
  Rt[(ij * kC + c) * kLp + (loff + l)] = f2bf(v);
}

__global__ void zero_rt_tail(unsigned short* __restrict__ Rt) {
  int t = blockIdx.x * blockDim.x + threadIdx.x;
  const int tail = kLp - kL;
  if (t >= 9 * kC * tail) return;
  int r = t % tail, rc = t / tail;
  Rt[rc * kLp + kL + r] = 0;
}

// L2-normalize each filter row, emit bf16 [kLp, 288] (pad rows zeroed)
__global__ void norm_wi(const float* __restrict__ wiF, unsigned short* __restrict__ wiN,
                        const float* __restrict__ esc) {
  int l = blockIdx.x, tid = threadIdx.x;
  __shared__ float red[128];
  if (l >= kL) {
    for (int k = tid; k < kK; k += 128) wiN[l * kK + k] = 0;
    return;
  }
  const float* row = wiF + l * kK;
  float s = 0.f;
  for (int k = tid; k < kK; k += 128) { float v = row[k]; s += v * v; }
  red[tid] = s; __syncthreads();
  for (int off = 64; off > 0; off >>= 1) {
    if (tid < off) red[tid] += red[tid + off];
    __syncthreads();
  }
  float inv = 1.f / fmaxf(sqrtf(red[0]), esc[0]);
  for (int k = tid; k < kK; k += 128) wiN[l * kK + k] = f2bf(row[k] * inv);
}

union FragBF { v16bf v; uint4 q[2]; unsigned short s[16]; };
union FragF  { v8f  v; float f[8]; };

// score[Lp, HW] = wiN[Lp, 288] x pmb[288, HW]   (bf16 WMMA, f32 acc)
__global__ __launch_bounds__(128)
void score_gemm(const unsigned short* __restrict__ A,
                const unsigned short* __restrict__ Bm,
                float* __restrict__ Cm) {
  int lane  = threadIdx.x;
  int mtile = blockIdx.y * 4 + threadIdx.y;   // 0..467
  int ntile = blockIdx.x;                     // 0..143
  int row = mtile * 16 + (lane & 15);
  int off = (lane >> 4) << 3;                 // K half-select per ISA A layout
  FragF acc;
#pragma unroll
  for (int r = 0; r < 8; ++r) acc.f[r] = 0.f;
#pragma unroll
  for (int kk = 0; kk < kK; kk += 32) {
    FragBF a, b;
    const unsigned short* ap = A + row * kK + kk + off;
    a.q[0] = *(const uint4*)(ap);
    a.q[1] = *(const uint4*)(ap + 16);
    const unsigned short* bp = Bm + (kk + off) * kHW + ntile * 16 + (lane & 15);
#pragma unroll
    for (int e = 0; e < 16; ++e) b.s[e] = bp[(e + (e & 8)) * kHW];
    acc.v = __builtin_amdgcn_wmma_f32_16x16x32_bf16(false, a.v, false, b.v,
                                                    (short)0, acc.v, false, false);
  }
  int ncol  = ntile * 16 + (lane & 15);
  int mbase = mtile * 16 + ((lane >> 4) << 3);
#pragma unroll
  for (int r = 0; r < 8; ++r) Cm[(mbase + r) * kHW + ncol] = acc.f[r];
}

__global__ void col_max(const float* __restrict__ S, float* __restrict__ cmax) {
  int p = blockIdx.x, tid = threadIdx.x;
  __shared__ float red[256];
  float m = -3.0e38f;
  for (int l = tid; l < kL; l += 256) m = fmaxf(m, S[l * kHW + p]);
  red[tid] = m; __syncthreads();
  for (int off = 128; off > 0; off >>= 1) {
    if (tid < off) red[tid] = fmaxf(red[tid], red[tid + off]);
    __syncthreads();
  }
  if (tid == 0) cmax[p] = red[0];
}

__global__ void col_sum(const float* __restrict__ S, const float* __restrict__ cmax,
                        float* __restrict__ csum) {
  int p = blockIdx.x, tid = threadIdx.x;
  __shared__ float red[256];
  float m = cmax[p], s = 0.f;
  for (int l = tid; l < kL; l += 256) s += __expf(kSm * (S[l * kHW + p] - m));
  red[tid] = s; __syncthreads();
  for (int off = 128; off > 0; off >>= 1) {
    if (tid < off) red[tid] += red[tid + off];
    __syncthreads();
  }
  if (tid == 0) csum[p] = red[0];
}

// attn halo: [kLp][50][50] bf16, zero border + zero pad rows
__global__ void build_attn(const float* __restrict__ S, const float* __restrict__ cmax,
                           const float* __restrict__ csum, unsigned short* __restrict__ at) {
  int t = blockIdx.x * blockDim.x + threadIdx.x;
  if (t >= kLp * kHalo) return;
  int pos = t % kHalo, l = t / kHalo;
  int yy = pos / kHaloW, xx = pos % kHaloW;
  unsigned short v = 0;
  if (l < kL && yy >= 1 && yy <= kH && xx >= 1 && xx <= kWd) {
    int p = (yy - 1) * kWd + (xx - 1);
    v = f2bf(__expf(kSm * (S[l * kHW + p] - cmax[p])) / csum[p]);
  }
  at[t] = v;
}

__global__ void zero_f32(float* __restrict__ p, int n) {
  int t = blockIdx.x * blockDim.x + threadIdx.x;
  if (t < n) p[t] = 0.f;
}

// out_acc[64, HW] += sum_ij Rt[ij][64, Lp] x attn_shift(ij)[Lp, HW]
// K split 6 ways across blockIdx.y (234 chunks of 32 -> 6 x 39).
// Per K-chunk, the attention halo footprint (32 rows x 3 y-rows x 18 cols)
// is staged into LDS once per block via async global->LDS dword copies
// (ASYNCcnt / s_wait_asynccnt), then all 4 waves x 9 taps read it from LDS.
__global__ __launch_bounds__(128)
void out_gemm(const unsigned short* __restrict__ Rt,
              const unsigned short* __restrict__ at,
              float* __restrict__ accbuf) {
  __shared__ unsigned short T[32][3][24];   // 4.6 KB staging tile
  int lane = threadIdx.x;
  int mt   = threadIdx.y;            // c-tile 0..3
  int tid  = mt * 32 + lane;
  int nt   = blockIdx.x;             // 0..143
  int y  = nt / 3;
  int x0 = (nt % 3) * 16;
  int off  = (lane >> 4) << 3;
  int arow = mt * 16 + (lane & 15);
  int kk0  = blockIdx.y * 39 * 32;
  FragF acc;
#pragma unroll
  for (int r = 0; r < 8; ++r) acc.f[r] = 0.f;
  for (int kc = 0; kc < 39; ++kc) {
    int kk = kk0 + kc * 32;
    __syncthreads();   // previous chunk's readers done before overwrite
    // 32 rows x 3 y-rows x 9 dwords = 864 async dword copies, 4B aligned
    // (kHalo=2500, kHaloW=50, x0 all even).
    for (int t = tid; t < 864; t += 128) {
      int d = t % 9, seg = t / 9;
      int r = seg / 3, i = seg % 3;
      int g = (kk + r) * kHalo + (y + i) * kHaloW + x0 + 2 * d;
      unsigned lds = (unsigned)(size_t)&T[r][i][2 * d];
      unsigned long long ga = (unsigned long long)(const void*)(at + g);
      asm volatile("global_load_async_to_lds_b32 %0, %1, off"
                   :: "v"(lds), "v"(ga) : "memory");
    }
    asm volatile("s_wait_asynccnt 0x0" ::: "memory");
    __syncthreads();
#pragma unroll
    for (int ij = 0; ij < 9; ++ij) {
      int i = ij / 3, j = ij % 3;
      FragBF a, b;
      const unsigned short* ap = Rt + (ij * kC + arow) * kLp + kk + off;
      a.q[0] = *(const uint4*)(ap);
      a.q[1] = *(const uint4*)(ap + 16);
#pragma unroll
      for (int e = 0; e < 16; ++e) b.s[e] = T[off + e + (e & 8)][i][j + (lane & 15)];
      acc.v = __builtin_amdgcn_wmma_f32_16x16x32_bf16(false, a.v, false, b.v,
                                                      (short)0, acc.v, false, false);
    }
  }
  int xcol  = x0 + (lane & 15);
  int cbase = mt * 16 + ((lane >> 4) << 3);
#pragma unroll
  for (int r = 0; r < 8; ++r)
    atomicAdd(&accbuf[((cbase + r) * kH + y) * kWd + xcol], acc.f[r]);
}

__global__ void finalize(const float* __restrict__ accbuf, const float* __restrict__ xin,
                         float* __restrict__ yout) {
  int t = blockIdx.x * blockDim.x + threadIdx.x;
  if (t >= kC * kHW) return;
  yout[t] = accbuf[t] * 0.25f + xin[t];
}

static inline int cdiv(int a, int b) { return (a + b - 1) / b; }

extern "C" void kernel_launch(void* const* d_in, const int* in_sizes, int n_in,
                              void* d_out, int out_size, void* d_ws, size_t ws_size,
                              hipStream_t stream) {
  (void)in_sizes; (void)n_in; (void)out_size; (void)ws_size;
  const float* x    = (const float*)d_in[0];
  const float* w_mb = (const float*)d_in[1];
  const float* b_mb = (const float*)d_in[2];
  const float* a_mb = (const float*)d_in[3];
  const float* w_m  = (const float*)d_in[4];
  const float* b_m  = (const float*)d_in[5];
  const float* a_m  = (const float*)d_in[6];
  const float* w_a  = (const float*)d_in[7];
  const float* b_a  = (const float*)d_in[8];
  const float* a_a  = (const float*)d_in[9];
  const float* esc  = (const float*)d_in[10];
  float* out = (float*)d_out;

  char* ws = (char*)d_ws;
  auto alloc = [&](size_t bytes) -> char* {
    char* p = ws;
    ws += (bytes + 255) & ~(size_t)255;
    return p;
  };
  float* rm    = (float*)alloc((size_t)4 * 43 * 48 * 4);
  float* mb    = (float*)alloc((size_t)kC2 * kHW * 4);
  float* tmpR  = (float*)alloc((size_t)kC * 43 * 48 * 4);
  float* refb  = (float*)alloc((size_t)kC * 43 * 43 * 4);
  float* fm    = (float*)alloc((size_t)kC2 * kHW * 4);
  float* fa    = (float*)alloc((size_t)kC * kHW * 4);
  float* wiF   = (float*)alloc((size_t)kL * kK * 4);
  unsigned short* wiN   = (unsigned short*)alloc((size_t)kLp * kK * 2);
  unsigned short* pmb   = (unsigned short*)alloc((size_t)kK * kHW * 2);
  float* score = (float*)alloc((size_t)kLp * kHW * 4);
  float* cmax  = (float*)alloc((size_t)kHW * 4);
  float* csum  = (float*)alloc((size_t)kHW * 4);
  unsigned short* attnH = (unsigned short*)alloc((size_t)kLp * kHalo * 2);
  unsigned short* Rt    = (unsigned short*)alloc((size_t)9 * kC * kLp * 2);
  float* accb  = (float*)alloc((size_t)kC * kHW * 4);

  build_resize_mats<<<dim3(cdiv(43, 64), 4), 64, 0, stream>>>(rm);

  const int hss[5]  = {48, 43, 38, 33, 28};
  const int loff[5] = {0, 2304, 4153, 5597, 6686};

  for (int b = 0; b < kNB; ++b) {
    const float* xb = x + (size_t)b * kC * kHW;
    float* yb = out + (size_t)b * kC * kHW;

    conv1x1_prelu<<<cdiv(kC2 * kHW, 256), 256, 0, stream>>>(xb, w_mb, b_mb, a_mb, mb,
                                                            kC, kC2, kHW);
    build_pmb<<<cdiv(kK * kHW, 256), 256, 0, stream>>>(mb, pmb);
    zero_rt_tail<<<cdiv(9 * kC * (kLp - kL), 256), 256, 0, stream>>>(Rt);

    for (int s = 0; s < 5; ++s) {
      int n = hss[s], npix = n * n;
      const float* refp;
      if (s == 0) {
        refp = xb;
      } else {
        const float* M = rm + (s - 1) * 43 * 48;
        resize_rows<<<cdiv(kC * n * 48, 256), 256, 0, stream>>>(xb, M, tmpR, n);
        resize_cols<<<cdiv(kC * n * n, 256), 256, 0, stream>>>(tmpR, M, refb, n);
        refp = refb;
      }
      conv1x1_prelu<<<cdiv(kC2 * npix, 256), 256, 0, stream>>>(refp, w_m, b_m, a_m, fm,
                                                               kC, kC2, npix);
      conv1x1_prelu<<<cdiv(kC * npix, 256), 256, 0, stream>>>(refp, w_a, b_a, a_a, fa,
                                                              kC, kC, npix);
      build_wi<<<cdiv(npix * kK, 256), 256, 0, stream>>>(fm, wiF, n, loff[s]);
      build_rt<<<cdiv(9 * kC * npix, 256), 256, 0, stream>>>(fa, Rt, n, loff[s]);
    }

    norm_wi<<<kLp, 128, 0, stream>>>(wiF, wiN, esc);
    score_gemm<<<dim3(kHW / 16, kLp / 64), dim3(32, 4), 0, stream>>>(wiN, pmb, score);
    col_max<<<kHW, 256, 0, stream>>>(score, cmax);
    col_sum<<<kHW, 256, 0, stream>>>(score, cmax, csum);
    build_attn<<<cdiv(kLp * kHalo, 256), 256, 0, stream>>>(score, cmax, csum, attnH);
    zero_f32<<<cdiv(kC * kHW, 256), 256, 0, stream>>>(accb, kC * kHW);
    out_gemm<<<dim3(144, 6), dim3(32, 4), 0, stream>>>(Rt, attnH, accb);
    finalize<<<cdiv(kC * kHW, 256), 256, 0, stream>>>(accb, xb, yb);
  }
}
